// InvResBlock_62113817035229
// MI455X (gfx1250) — matrix operand order repaired and checked by reading, not compiled, not run on hardware
//
#include <hip/hip_runtime.h>

typedef _Float16 h8   __attribute__((ext_vector_type(8)));
typedef _Float16 v16h __attribute__((ext_vector_type(16)));
typedef float    v8f  __attribute__((ext_vector_type(8)));

#define B_ROWS 16384
#define C_DIM  1024
#define LDT    40          // LDS row stride in halves (80B, keeps 16B alignment)

#define M_ELU   0
#define M_OUT   1
#define M_MASK  2
#define M_PLAIN 3

// ---------------------------------------------------------------------------
// GEMM: out[b,i] = sum_j A[b,j] * Wh[i,j]   (A: B_ROWS x C f16, Wh: C x C f16)
// Staging: CDNA5 async copies (global_load_async_to_lds_b128, ASYNCcnt),
// 4 LDS buffers with 3 tiles in flight; in-order ASYNCcnt lets us wait for
// only the oldest tile (6 ops/tile -> s_wait_asynccnt 12).
// Restage is issued right after the barrier (before the WMMA block) to avoid
// WMMA->VALU WAR hazard NOPs on the address computation and to give the DMA
// engine the maximum head start.
// ---------------------------------------------------------------------------
template<int MODE>
__global__ __launch_bounds__(256)
void gemm_wmma(const _Float16* __restrict__ A,
               const _Float16* __restrict__ Wh,
               const float*    __restrict__ bias,
               const float*    __restrict__ Yadd,
               const _Float16* __restrict__ Dmask,
               _Float16*       __restrict__ HOut,
               _Float16*       __restrict__ DOut,
               float*          __restrict__ ZOut)
{
    __shared__ __attribute__((aligned(16))) _Float16 As[4][128 * LDT];
    __shared__ __attribute__((aligned(16))) _Float16 Bs[4][256 * LDT];

    const int tid  = threadIdx.x;
    const int lane = tid & 31;
    const int wv   = tid >> 5;
    const int wM   = wv & 1;     // 2 waves along M
    const int wN   = wv >> 1;    // 4 waves along N
    const int m0   = blockIdx.x * 128;
    const int n0   = blockIdx.y * 256;

    const int g = lane >> 4;     // lane group (0/1)
    const int r = lane & 15;

    // Per-thread staging addresses:
    //   A tile 128x32 halves: thread -> row ar, 16-half chunk ac (2 x 16B)
    //   B tile 256x32 halves: thread -> row tid (4 x 16B)
    const int ar = tid >> 1;
    const int ac = (tid & 1) << 4;
    const _Float16* gA = A  + (size_t)(m0 + ar) * C_DIM + ac;
    const _Float16* gB = Wh + (size_t)(n0 + tid) * C_DIM;

    // LDS byte offsets (flat shared address truncates to LDS offset)
    unsigned ldsA[4], ldsB[4];
#pragma unroll
    for (int b = 0; b < 4; ++b) {
        ldsA[b] = (unsigned)(uintptr_t)&As[b][ar * LDT + ac];
        ldsB[b] = (unsigned)(uintptr_t)&Bs[b][tid * LDT];
    }

    v8f acc[4][4];
#pragma unroll
    for (int i = 0; i < 4; ++i)
#pragma unroll
        for (int j = 0; j < 4; ++j)
#pragma unroll
            for (int e = 0; e < 8; ++e) acc[i][j][e] = 0.f;

    // async copy of one K-step tile into LDS buffer `buf`
    // (INST_OFFSET applies to both LDS and global address; row chunks are
    //  contiguous on both sides, so offset:16/32/48 is consistent)
    auto stage_async = [&](int kt, int buf) {
        const unsigned long long pa = (unsigned long long)(gA + kt * 32);
        const unsigned long long pb = (unsigned long long)(gB + kt * 32);
        const unsigned la = ldsA[buf];
        const unsigned lb = ldsB[buf];
        asm volatile("global_load_async_to_lds_b128 %0, %1, off"
                     :: "v"(la), "v"(pa) : "memory");
        asm volatile("global_load_async_to_lds_b128 %0, %1, off offset:16"
                     :: "v"(la), "v"(pa) : "memory");
        asm volatile("global_load_async_to_lds_b128 %0, %1, off"
                     :: "v"(lb), "v"(pb) : "memory");
        asm volatile("global_load_async_to_lds_b128 %0, %1, off offset:16"
                     :: "v"(lb), "v"(pb) : "memory");
        asm volatile("global_load_async_to_lds_b128 %0, %1, off offset:32"
                     :: "v"(lb), "v"(pb) : "memory");
        asm volatile("global_load_async_to_lds_b128 %0, %1, off offset:48"
                     :: "v"(lb), "v"(pb) : "memory");
    };

    // prologue: 3 tiles in flight
    stage_async(0, 0);
    stage_async(1, 1);
    stage_async(2, 2);

#pragma unroll 4
    for (int kt = 0; kt < 32; ++kt) {
        // wait for the oldest tile only (<= 2 tiles = 12 ops may remain)
        asm volatile("s_wait_asynccnt 0xc" ::: "memory");
        __syncthreads();

        // restage buffer (kt-1)&3 immediately: all waves drained their reads
        // of it before signaling the barrier we just passed.
        if (kt < 29) stage_async(kt + 3, (kt + 3) & 3);

        const int cur = kt & 3;
        const _Float16* Ac = As[cur];
        const _Float16* Bc = Bs[cur];

        // Preload fragments (ds_load_b128), then 16 WMMAs.
        // B fragments: 32(K) x 16(N); lane: col = r, halves -> K = i + 16*g
        v16h bf[4];
#pragma unroll
        for (int nb = 0; nb < 4; ++nb) {
            const _Float16* p = Bc + (wN * 64 + nb * 16 + r) * LDT + (g << 4);
            union { v16h v; h8 h[2]; } u;
            u.h[0] = *(const h8*)p;
            u.h[1] = *(const h8*)(p + 8);
            bf[nb] = u.v;
        }
        // A fragments: 16(M) x 32(K); lane: row = r, halves -> K = i + 16*g
        v16h af[4];
#pragma unroll
        for (int ma = 0; ma < 4; ++ma) {
            const _Float16* p = Ac + (wM * 64 + ma * 16 + r) * LDT + (g << 4);
            union { v16h v; h8 h[2]; } u;
            u.h[0] = *(const h8*)p;
            u.h[1] = *(const h8*)(p + 8);
            af[ma] = u.v;
        }
#pragma unroll
        for (int ma = 0; ma < 4; ++ma)
#pragma unroll
            for (int nb = 0; nb < 4; ++nb)
                acc[ma][nb] = __builtin_amdgcn_wmma_f32_16x16x32_f16(
                    false, af[ma], false, bf[nb], (short)0, acc[ma][nb], false, false);
    }

    // Epilogue: C/D layout -> lane: N = r, element e: M = e + 8*g
#pragma unroll
    for (int ma = 0; ma < 4; ++ma) {
#pragma unroll
        for (int nb = 0; nb < 4; ++nb) {
            const int col  = n0 + wN * 64 + nb * 16 + r;
            const int rowb = m0 + wM * 64 + ma * 16 + g * 8;
            float bcol = 0.f;
            if (MODE == M_ELU || MODE == M_OUT) bcol = bias[col];
#pragma unroll
            for (int e = 0; e < 8; ++e) {
                const size_t idx = (size_t)(rowb + e) * C_DIM + col;
                float val = acc[ma][nb][e];
                if (MODE == M_ELU) {
                    float t = val + bcol;
                    float hh, dd;
                    if (t > 0.f) { hh = t; dd = 1.f; }
                    else { float ex = __expf(t); hh = ex - 1.f; dd = ex; }
                    HOut[idx] = (_Float16)hh;
                    DOut[idx] = (_Float16)dd;
                } else if (MODE == M_OUT) {
                    ZOut[idx] = val + bcol + Yadd[idx];
                } else if (MODE == M_MASK) {
                    HOut[idx] = (_Float16)(val * (float)Dmask[idx]);
                } else {
                    HOut[idx] = (_Float16)val;
                }
            }
        }
    }
}

// Convert f32 weights to f16 (row-major) and f16-transposed copies.
__global__ void cvt_weights(const float* __restrict__ W1, const float* __restrict__ W2,
                            const float* __restrict__ W3,
                            _Float16* __restrict__ Wf, _Float16* __restrict__ Wt)
{
    const int idx = blockIdx.x * 256 + threadIdx.x;   // over 1024*1024
    const int i = idx >> 10, j = idx & 1023;
    const float* Ws[3] = { W1, W2, W3 };
#pragma unroll
    for (int m = 0; m < 3; ++m) {
        float x = Ws[m][idx];
        Wf[(size_t)m * (C_DIM * C_DIM) + idx]                   = (_Float16)x;
        Wt[(size_t)m * (C_DIM * C_DIM) + (size_t)j * C_DIM + i] = (_Float16)x;
    }
}

// f32 -> f16 activation convert; optionally zero the per-row accumulator.
__global__ void cvt_act(const float* __restrict__ src, _Float16* __restrict__ dst,
                        float* __restrict__ accb, size_t n)
{
    const size_t i = (size_t)blockIdx.x * 256 + threadIdx.x;
    if (i < n) dst[i] = (_Float16)src[i];
    if (accb != nullptr && i < B_ROWS) accb[i] = 0.f;
}

// acc[b] += coef * sum_c w[b,c]*v[b,c]
__global__ void reduce_row(const _Float16* __restrict__ w, const float* __restrict__ v,
                           float* __restrict__ accb, float coef)
{
    const int b = blockIdx.x;
    const _Float16* wr = w + (size_t)b * C_DIM;
    const float*    vr = v + (size_t)b * C_DIM;
    float s = 0.f;
    for (int c = threadIdx.x; c < C_DIM; c += 256)
        s += (float)wr[c] * vr[c];
    __shared__ float red[256];
    red[threadIdx.x] = s;
    __syncthreads();
    for (int st = 128; st > 0; st >>= 1) {
        if (threadIdx.x < st) red[threadIdx.x] += red[threadIdx.x + st];
        __syncthreads();
    }
    if (threadIdx.x == 0) accb[b] += coef * red[0];
}

__global__ void finalize(const float* __restrict__ ldj, const float* __restrict__ accb,
                         float* __restrict__ outLd)
{
    const int b = blockIdx.x * 256 + threadIdx.x;
    if (b < B_ROWS) outLd[b] = ldj[b] + accb[b];
}

extern "C" void kernel_launch(void* const* d_in, const int* in_sizes, int n_in,
                              void* d_out, int out_size, void* d_ws, size_t ws_size,
                              hipStream_t stream)
{
    const float* y   = (const float*)d_in[0];
    const float* ldj = (const float*)d_in[1];
    const float* v   = (const float*)d_in[2];
    const float* W1  = (const float*)d_in[3];
    const float* b1  = (const float*)d_in[4];
    const float* W2  = (const float*)d_in[5];
    const float* b2  = (const float*)d_in[6];
    const float* W3  = (const float*)d_in[7];
    const float* b3  = (const float*)d_in[8];

    float* z   = (float*)d_out;                               // B_ROWS*C_DIM
    float* oLd = (float*)d_out + (size_t)B_ROWS * C_DIM;      // B_ROWS

    const size_t WELEM = (size_t)C_DIM * C_DIM;   // 1M
    const size_t AELEM = (size_t)B_ROWS * C_DIM;  // 16M

    _Float16* Wf   = (_Float16*)d_ws;          // 3 x WELEM f16 (row-major)
    _Float16* Wt   = Wf + 3 * WELEM;           // 3 x WELEM f16 (transposed)
    _Float16* D1   = Wt + 3 * WELEM;           // AELEM f16 elu'(a1)
    _Float16* D2   = D1 + AELEM;               // AELEM f16 elu'(a2)
    _Float16* P0   = D2 + AELEM;               // AELEM f16 ping
    _Float16* P1   = P0 + AELEM;               // AELEM f16 pong
    float*    accb = (float*)(P1 + AELEM);     // B_ROWS f32

    dim3 blk(256);
    dim3 gG(B_ROWS / 128, C_DIM / 256);

    // weight conversion (+transpose), y->f16, zero log-det accumulator
    cvt_weights<<<dim3(WELEM / 256), blk, 0, stream>>>(W1, W2, W3, Wf, Wt);
    cvt_act<<<dim3((AELEM + 255) / 256), blk, 0, stream>>>(y, P0, accb, AELEM);

    // forward: h1,d1 ; h2,d2 ; z = y + g(y)
    gemm_wmma<M_ELU><<<gG, blk, 0, stream>>>(P0, Wf,             b1, nullptr, nullptr, P1, D1, nullptr);
    gemm_wmma<M_ELU><<<gG, blk, 0, stream>>>(P1, Wf + WELEM,     b2, nullptr, nullptr, P0, D2, nullptr);
    gemm_wmma<M_OUT><<<gG, blk, 0, stream>>>(P0, Wf + 2 * WELEM, b3, y,       nullptr, nullptr, nullptr, z);

    // w0 = v
    cvt_act<<<dim3((AELEM + 255) / 256), blk, 0, stream>>>(v, P0, nullptr, AELEM);

    _Float16* wc = P0;
    _Float16* tp = P1;
    for (int k = 1; k <= 16; ++k) {
        // w <- J^T w : ((w @ W3) .* d2 @ W2) .* d1 @ W1
        gemm_wmma<M_MASK ><<<gG, blk, 0, stream>>>(wc, Wt + 2 * WELEM, nullptr, nullptr, D2, tp, nullptr, nullptr);
        gemm_wmma<M_MASK ><<<gG, blk, 0, stream>>>(tp, Wt + WELEM,     nullptr, nullptr, D1, wc, nullptr, nullptr);
        gemm_wmma<M_PLAIN><<<gG, blk, 0, stream>>>(wc, Wt,             nullptr, nullptr, nullptr, tp, nullptr, nullptr);
        const float coef = ((k & 1) ? 1.f : -1.f) / (float)k;
        reduce_row<<<dim3(B_ROWS), blk, 0, stream>>>(tp, v, accb, coef);
        _Float16* s = wc; wc = tp; tp = s;
    }

    finalize<<<dim3((B_ROWS + 255) / 256), blk, 0, stream>>>(ldj, accb, oLd);
}
